// LSTM_15444702397091
// MI455X (gfx1250) — matrix operand order repaired and checked by reading, not compiled, not run on hardware
//
#include <hip/hip_runtime.h>

// ---------------------------------------------------------------------------
// 2-layer LSTM (B=2048, T=512, IN=60, H=128) + FC(128->1) for gfx1250.
// bf16 WMMA (v_wmma_f32_16x16x32_bf16); weights LDS-resident, swizzled into
// per-wave-contiguous 1KB fragments (all B loads = ds_load_b128 with
// immediate offsets). Activation tile [x|h] double-buffered in LDS => one
// barrier per timestep. Cell state in VGPRs for all 512 steps. All 4 gate
// B-fragments of a k-tile are loaded before their WMMAs so the scheduler can
// use a partial s_wait_dscnt ladder. Activations use hardware V_TANH_F32
// when available (sigmoid via tanh identity) => 5 TRANS ops instead of 10.
// ---------------------------------------------------------------------------

typedef __attribute__((ext_vector_type(16))) __bf16 v16bf;
typedef __attribute__((ext_vector_type(8)))  float  v8f;

#define Bn   2048
#define Tn   512
#define INF_ 60
#define Hn   128

union AFrag { v16bf v; uint4 q[2]; };
static_assert(sizeof(AFrag) == 32, "frag size");

__device__ __forceinline__ float hw_tanh(float x) {
#if __has_builtin(__builtin_amdgcn_tanhf)
    return __builtin_amdgcn_tanhf(x);
#elif __has_builtin(__builtin_amdgcn_tanh_f32)
    return __builtin_amdgcn_tanh_f32(x);
#else
    return 1.0f - 2.0f / (__expf(2.0f * x) + 1.0f);
#endif
}
__device__ __forceinline__ float fast_sigmoid(float x) {
    // sigmoid(x) = 0.5 + 0.5 * tanh(x/2): one TRANS op on CDNA5
    return fmaf(0.5f, hw_tanh(0.5f * x), 0.5f);
}

#define WMMA_BF16(A, B, C) \
    __builtin_amdgcn_wmma_f32_16x16x32_bf16(false, (A), false, (B), (short)0, (C), false, false)

// Load all 4 gate B-fragments of k-tile `kt`, then issue the 4 WMMAs.
#define KTILE_STEP(kt)                                                        \
    {                                                                         \
        AFrag a;                                                              \
        a.q[0] = *(const uint4*)(ar + (kt) * 32);                             \
        a.q[1] = *(const uint4*)(ar + (kt) * 32 + 16);                        \
        AFrag b0, b1, b2, b3;                                                 \
        b0.q[0] = *(const uint4*)(bw_base + ((kt) * 4 + 0) * 1024);           \
        b0.q[1] = *(const uint4*)(bw_base + ((kt) * 4 + 0) * 1024 + 16);      \
        b1.q[0] = *(const uint4*)(bw_base + ((kt) * 4 + 1) * 1024);           \
        b1.q[1] = *(const uint4*)(bw_base + ((kt) * 4 + 1) * 1024 + 16);      \
        b2.q[0] = *(const uint4*)(bw_base + ((kt) * 4 + 2) * 1024);           \
        b2.q[1] = *(const uint4*)(bw_base + ((kt) * 4 + 2) * 1024 + 16);      \
        b3.q[0] = *(const uint4*)(bw_base + ((kt) * 4 + 3) * 1024);           \
        b3.q[1] = *(const uint4*)(bw_base + ((kt) * 4 + 3) * 1024 + 16);      \
        ai = WMMA_BF16(a.v, b0.v, ai);                                        \
        af = WMMA_BF16(a.v, b1.v, af);                                        \
        ag = WMMA_BF16(a.v, b2.v, ag);                                        \
        ao = WMMA_BF16(a.v, b3.v, ao);                                        \
    }

// ---------------------------------------------------------------------------
// Layer 0: x[B,T,60] (fp32) -> hs0[T,B,128] (bf16).  K = 64(pad x) + 128(h).
// ---------------------------------------------------------------------------
__global__ __launch_bounds__(256) void lstm_layer0(
    const float* __restrict__ x,  const float* __restrict__ Wx,
    const float* __restrict__ Wh, const float* __restrict__ bx,
    const float* __restrict__ bh, __bf16* __restrict__ hs0)
{
    constexpr int KT    = 6;        // k-tiles of 32: 2 (x, padded 64) + 4 (h)
    constexpr int KLD   = KT * 32;  // 192
    constexpr int KXP   = 64;
    constexpr int NFRAG = KT * 32;  // 192 fragments (1 KB each)

    extern __shared__ char smem[];
    __bf16* wlds  = (__bf16*)smem;                               // 192 KB
    float*  biasL = (float*)(smem + NFRAG * 1024);               // 2 KB
    __bf16* xbuf0 = (__bf16*)(smem + NFRAG * 1024 + 2048);       // 6 KB
    __bf16* xbuf1 = xbuf0 + 16 * KLD;                            // 6 KB

    const int tid = threadIdx.x;
    const int b0  = blockIdx.x * 16;

    // swizzled bf16 weights, per-wave contiguous; frag-in-wave = kt*4 + gate
    for (int e = tid; e < NFRAG * 512; e += 256) {
        int frag = e >> 9, rem = e & 511;
        int L  = rem >> 4, j = rem & 15;
        int w_ = frag / (KT * 4);
        int r2 = frag % (KT * 4);
        int kt = r2 >> 2, g = r2 & 3;
        int nt = g * 8 + w_;
        int k  = kt * 32 + ((L >> 4) << 3) + (j & 7) + ((j >> 3) << 4);
        int n  = nt * 16 + (L & 15);
        float v = 0.f;
        if (k < INF_)       v = Wx[n * INF_ + k];
        else if (k >= KXP)  v = Wh[n * Hn + (k - KXP)];
        wlds[e] = (__bf16)v;
    }
    for (int e = tid; e < 512; e += 256) biasL[e] = bx[e] + bh[e];
    for (int e = tid; e < 2 * 16 * KLD; e += 256) xbuf0[e] = (__bf16)0.f;

    // pre-stage x(0)
    #pragma unroll
    for (int j = 0; j < 4; ++j) {
        int idx = tid + j * 256;
        int row = idx >> 6, col = idx & 63;
        float v = (col < INF_) ? x[((size_t)(b0 + row) * Tn) * INF_ + col] : 0.f;
        xbuf0[row * KLD + col] = (__bf16)v;
    }
    __syncthreads();

    const int w    = tid >> 5;
    const int L    = tid & 31;
    const int ln   = L & 15;
    const int half = L >> 4;
    const float bi  = biasL[      w * 16 + ln];
    const float bff = biasL[128 + w * 16 + ln];
    const float bg  = biasL[256 + w * 16 + ln];
    const float bo  = biasL[384 + w * 16 + ln];
    const int hc = w * 16 + ln;
    const int crow = tid >> 4, cchunk = tid & 15;
    const char* bw_base = (const char*)wlds + w * (KT * 4 * 1024) + L * 32;

    float c[8];
    #pragma unroll
    for (int r = 0; r < 8; ++r) c[r] = 0.f;

    __bf16* rb = xbuf0;
    __bf16* wb = xbuf1;

    #pragma unroll 1
    for (int t = 0; t < Tn; ++t) {
        // issue next step's global x loads early
        float sx[4];
        if (t + 1 < Tn) {
            #pragma unroll
            for (int j = 0; j < 4; ++j) {
                int idx = tid + j * 256;
                int row = idx >> 6, col = idx & 63;
                sx[j] = (col < INF_)
                    ? x[((size_t)(b0 + row) * Tn + (t + 1)) * INF_ + col] : 0.f;
            }
        }
        // pipelined coalesced copy-out of h(t-1)
        if (t > 0) {
            uint4 hq = *((const uint4*)(rb + crow * KLD + KXP) + cchunk);
            *((uint4*)(hs0 + ((size_t)(t - 1) * Bn + b0 + crow) * Hn) + cchunk) = hq;
        }

        // WMMA: gates[16,512] = [x|h][16,192] @ W[192,512]
        const __bf16* ar = rb + ln * KLD + 8 * half;
        v8f ai = {}, af = {}, ag = {}, ao = {};
        #pragma unroll
        for (int kt = 0; kt < KT; ++kt)
            KTILE_STEP(kt)

        // elementwise LSTM cell update; h(t) -> write buffer
        #pragma unroll
        for (int r = 0; r < 8; ++r) {
            float iv = fast_sigmoid(ai[r] + bi);
            float fv = fast_sigmoid(af[r] + bff);
            float gv = hw_tanh     (ag[r] + bg);
            float ov = fast_sigmoid(ao[r] + bo);
            c[r] = fv * c[r] + iv * gv;
            float hv = ov * hw_tanh(c[r]);
            int row = r + 8 * half;
            wb[row * KLD + KXP + hc] = (__bf16)hv;
        }
        // staged x(t+1) -> write buffer
        if (t + 1 < Tn) {
            #pragma unroll
            for (int j = 0; j < 4; ++j) {
                int idx = tid + j * 256;
                int row = idx >> 6, col = idx & 63;
                wb[row * KLD + col] = (__bf16)sx[j];
            }
        }
        __syncthreads();
        __bf16* tmp = rb; rb = wb; wb = tmp;
    }

    // final copy-out: h(T-1) sits in rb after the last swap
    uint4 hq = *((const uint4*)(rb + crow * KLD + KXP) + cchunk);
    *((uint4*)(hs0 + ((size_t)(Tn - 1) * Bn + b0 + crow) * Hn) + cchunk) = hq;
}

// ---------------------------------------------------------------------------
// Layer 1: hs0[T,B,128] (bf16) -> h1f[B,128] (fp32, last step).  K = 256.
// ---------------------------------------------------------------------------
__global__ __launch_bounds__(256) void lstm_layer1(
    const __bf16* __restrict__ hs0, const float* __restrict__ Wx,
    const float* __restrict__ Wh,   const float* __restrict__ bx,
    const float* __restrict__ bh,   float* __restrict__ h1f)
{
    constexpr int KT    = 8;
    constexpr int KLD   = KT * 32;  // 256
    constexpr int KXP   = 128;
    constexpr int NFRAG = KT * 32;  // 256 fragments

    extern __shared__ char smem[];
    __bf16* wlds  = (__bf16*)smem;                               // 256 KB
    float*  biasL = (float*)(smem + NFRAG * 1024);               // 2 KB
    __bf16* xbuf0 = (__bf16*)(smem + NFRAG * 1024 + 2048);       // 8 KB
    __bf16* xbuf1 = xbuf0 + 16 * KLD;                            // 8 KB

    const int tid = threadIdx.x;
    const int b0  = blockIdx.x * 16;

    for (int e = tid; e < NFRAG * 512; e += 256) {
        int frag = e >> 9, rem = e & 511;
        int L  = rem >> 4, j = rem & 15;
        int w_ = frag / (KT * 4);
        int r2 = frag % (KT * 4);
        int kt = r2 >> 2, g = r2 & 3;
        int nt = g * 8 + w_;
        int k  = kt * 32 + ((L >> 4) << 3) + (j & 7) + ((j >> 3) << 4);
        int n  = nt * 16 + (L & 15);
        float v = (k < KXP) ? Wx[n * Hn + k] : Wh[n * Hn + (k - KXP)];
        wlds[e] = (__bf16)v;
    }
    for (int e = tid; e < 512; e += 256) biasL[e] = bx[e] + bh[e];
    for (int e = tid; e < 2 * 16 * KLD; e += 256) xbuf0[e] = (__bf16)0.f;

    const int srow = tid >> 4, schunk = tid & 15;
    {   // pre-stage h0(0)
        uint4 q = *((const uint4*)(hs0 + ((size_t)(b0 + srow)) * Hn) + schunk);
        *((uint4*)(xbuf0 + srow * KLD) + schunk) = q;
    }
    __syncthreads();

    const int w    = tid >> 5;
    const int L    = tid & 31;
    const int ln   = L & 15;
    const int half = L >> 4;
    const float bi  = biasL[      w * 16 + ln];
    const float bff = biasL[128 + w * 16 + ln];
    const float bg  = biasL[256 + w * 16 + ln];
    const float bo  = biasL[384 + w * 16 + ln];
    const int hc = w * 16 + ln;
    const char* bw_base = (const char*)wlds + w * (KT * 4 * 1024) + L * 32;

    float c[8];
    #pragma unroll
    for (int r = 0; r < 8; ++r) c[r] = 0.f;

    __bf16* rb = xbuf0;
    __bf16* wb = xbuf1;

    #pragma unroll 1
    for (int t = 0; t < Tn; ++t) {
        // issue next step's h0 tile load early
        uint4 sq;
        if (t + 1 < Tn)
            sq = *((const uint4*)(hs0 + ((size_t)(t + 1) * Bn + b0 + srow) * Hn)
                   + schunk);

        // WMMA: gates[16,512] = [h0|h1][16,256] @ W[256,512]
        const __bf16* ar = rb + ln * KLD + 8 * half;
        v8f ai = {}, af = {}, ag = {}, ao = {};
        #pragma unroll
        for (int kt = 0; kt < KT; ++kt)
            KTILE_STEP(kt)

        // elementwise cell update; h1(t) -> write buffer
        #pragma unroll
        for (int r = 0; r < 8; ++r) {
            float iv = fast_sigmoid(ai[r] + bi);
            float fv = fast_sigmoid(af[r] + bff);
            float gv = hw_tanh     (ag[r] + bg);
            float ov = fast_sigmoid(ao[r] + bo);
            c[r] = fv * c[r] + iv * gv;
            float hv = ov * hw_tanh(c[r]);
            int row = r + 8 * half;
            wb[row * KLD + KXP + hc] = (__bf16)hv;
            if (t == Tn - 1)
                h1f[(size_t)(b0 + row) * Hn + hc] = hv;
        }
        // staged h0(t+1) -> write buffer x-part
        if (t + 1 < Tn)
            *((uint4*)(wb + srow * KLD) + schunk) = sq;
        __syncthreads();
        __bf16* tmp = rb; rb = wb; wb = tmp;
    }
}

// ---------------------------------------------------------------------------
// Final FC: out[b] = h1f[b,:] . Wfc + bfc
// ---------------------------------------------------------------------------
__global__ __launch_bounds__(256) void fc_kernel(
    const float* __restrict__ h1, const float* __restrict__ Wfc,
    const float* __restrict__ bfc, float* __restrict__ out)
{
    int b = blockIdx.x * blockDim.x + threadIdx.x;
    if (b < Bn) {
        float s = bfc[0];
        #pragma unroll
        for (int j = 0; j < Hn; ++j)
            s += h1[(size_t)b * Hn + j] * Wfc[j];
        out[b] = s;
    }
}

extern "C" void kernel_launch(void* const* d_in, const int* in_sizes, int n_in,
                              void* d_out, int out_size, void* d_ws, size_t ws_size,
                              hipStream_t stream) {
    const float* x   = (const float*)d_in[0];
    const float* Wx0 = (const float*)d_in[1];
    const float* bx0 = (const float*)d_in[2];
    const float* Wh0 = (const float*)d_in[3];
    const float* bh0 = (const float*)d_in[4];
    const float* Wx1 = (const float*)d_in[5];
    const float* bx1 = (const float*)d_in[6];
    const float* Wh1 = (const float*)d_in[7];
    const float* bh1 = (const float*)d_in[8];
    const float* Wfc = (const float*)d_in[9];
    const float* bfc = (const float*)d_in[10];

    __bf16* hs0 = (__bf16*)d_ws;                                    // 268 MB
    float*  h1f = (float*)((char*)d_ws + (size_t)Tn * Bn * Hn * 2); // 1 MB

    const size_t smem0 = 192 * 1024 + 2048 + 2 * 16 * 192 * 2;  // ~206 KB
    const size_t smem1 = 256 * 1024 + 2048 + 2 * 16 * 256 * 2;  // ~274 KB
    (void)hipFuncSetAttribute((const void*)lstm_layer0,
        hipFuncAttributeMaxDynamicSharedMemorySize, (int)smem0);
    (void)hipFuncSetAttribute((const void*)lstm_layer1,
        hipFuncAttributeMaxDynamicSharedMemorySize, (int)smem1);

    lstm_layer0<<<dim3(Bn / 16), dim3(256), smem0, stream>>>(
        x, Wx0, Wh0, bx0, bh0, hs0);
    lstm_layer1<<<dim3(Bn / 16), dim3(256), smem1, stream>>>(
        hs0, Wx1, Wh1, bx1, bh1, h1f);
    fc_kernel<<<dim3(Bn / 256), dim3(256), 0, stream>>>(
        h1f, Wfc, bfc, (float*)d_out);
}
